// Gru4RecModel_34617436405927
// MI455X (gfx1250) — matrix operand order repaired and checked by reading, not compiled, not run on hardware
//
#include <hip/hip_runtime.h>
#include <math.h>
#include <stdint.h>

// Problem constants (from reference): B=256, T=200, H=256, V=50000
#define BB 256
#define TT 200
#define HH 256
#define VV 50000
#define G3 768          // 3*H
#define EMB_DW 6400128  // (V+1)*H/2 dwords of packed bf16
#define WP_DW  196608   // 16 ktiles * 48 ntiles * 32 lanes * 8 dwords

typedef __attribute__((ext_vector_type(16))) __bf16 v16bf;
typedef __attribute__((ext_vector_type(8)))  float  v8f;

struct u4_t { unsigned int a, b, c, d; };   // POD 128-bit chunk (union-safe)

union Frag {
  unsigned int u[8];
  u4_t         q[2];
  v16bf        v;
};

__device__ __forceinline__ unsigned short f2bf(float f) {
  unsigned int u = __float_as_uint(f);
  unsigned int r = u + 0x7fffu + ((u >> 16) & 1u);   // round-to-nearest-even
  return (unsigned short)(r >> 16);
}
__device__ __forceinline__ unsigned int pack2bf(float lo, float hi) {
  return (unsigned int)f2bf(lo) | ((unsigned int)f2bf(hi) << 16);
}

// Fast transcendentals on the native v_exp_f32 path (saturate correctly).
__device__ __forceinline__ float sigmoid_fast(float x) {
  return 1.0f / (1.0f + __expf(-x));
}
__device__ __forceinline__ float tanh_fast(float x) {
  return 1.0f - 2.0f / (__expf(2.0f * x) + 1.0f);
}

__device__ __forceinline__ v8f wmma_bf16(const Frag& a, const Frag& b, v8f c) {
  return __builtin_amdgcn_wmma_f32_16x16x32_bf16(false, a.v, false, b.v,
                                                 (short)0, c, false, false);
}

// ---------------------------------------------------------------------------
// Pack embedding table f32 -> bf16 pairs (row-major, 128 dwords per row).
// ---------------------------------------------------------------------------
__global__ void pack_emb_kernel(const float* __restrict__ emb,
                                unsigned int* __restrict__ out) {
  int i = blockIdx.x * blockDim.x + threadIdx.x;
  if (i < EMB_DW) out[i] = pack2bf(emb[2 * i], emb[2 * i + 1]);
}

// ---------------------------------------------------------------------------
// Pack [w_ih; w_hh] (512 x 768 f32) into WMMA B-fragment layout:
//   out[((kt*48 + nt)*32 + lane)*8 + d] = {W(k,n), W(k+1,n)} as bf16 pair
//   k = kt*32 + (lane>>4)*16 + 2d,  n = nt*16 + (lane&15)
// -> each lane's fragment is 32 contiguous bytes.
// ---------------------------------------------------------------------------
__global__ void pack_w_kernel(const float* __restrict__ w_ih,
                              const float* __restrict__ w_hh,
                              unsigned int* __restrict__ out) {
  int i = blockIdx.x * blockDim.x + threadIdx.x;
  if (i >= WP_DW) return;
  int d    = i & 7;
  int lane = (i >> 3) & 31;
  int rest = i >> 8;
  int nt   = rest % 48;
  int kt   = rest / 48;
  int k = kt * 32 + (lane >> 4) * 16 + d * 2;
  int n = nt * 16 + (lane & 15);
  float v0, v1;
  if (k < HH) { v0 = w_ih[k * G3 + n];        v1 = w_ih[(k + 1) * G3 + n]; }
  else        { v0 = w_hh[(k - HH) * G3 + n]; v1 = w_hh[(k - HH + 1) * G3 + n]; }
  out[i] = pack2bf(v0, v1);
}

// ---- helpers for the recurrence hot loop -----------------------------------
__device__ __forceinline__ void load_b6(Frag* f, const unsigned int* wpt,
                                        int kt, int c0, int lane) {
#pragma unroll
  for (int ct = 0; ct < 2; ++ct)
#pragma unroll
    for (int g = 0; g < 3; ++g) {
      const u4_t* p =
          (const u4_t*)(wpt + (((kt * 48) + g * 16 + c0 + ct) * 32 + lane) * 8);
      f[ct * 3 + g].q[0] = p[0];
      f[ct * 3 + g].q[1] = p[1];
    }
}

__device__ __forceinline__ void load_a(Frag& fa, const unsigned int* src,
                                       int nl, int half, int kt7) {
  int base = nl * 128 + kt7 * 16 + half * 8;
#pragma unroll
  for (int d = 0; d < 4; ++d) fa.u[d]     = src[base + d];
#pragma unroll
  for (int d = 0; d < 4; ++d) fa.u[4 + d] = src[base + 4 + d];
}

__device__ __forceinline__ void wmma6(const Frag& fa, Frag* fb, int kt,
                                      v8f* accz, v8f* accr,
                                      v8f* accxh, v8f* accrh) {
#pragma unroll
  for (int ct = 0; ct < 2; ++ct) {
    accz[ct] = wmma_bf16(fa, fb[ct * 3 + 0], accz[ct]);
    accr[ct] = wmma_bf16(fa, fb[ct * 3 + 1], accr[ct]);
    if (kt < 8) accxh[ct] = wmma_bf16(fa, fb[ct * 3 + 2], accxh[ct]);
    else        accrh[ct] = wmma_bf16(fa, fb[ct * 3 + 2], accrh[ct]);
  }
}

// ---------------------------------------------------------------------------
// GRU recurrence: 16 blocks x 256 threads (8 waves, 2 waves/SIMD). Block owns
// 16 batch rows. Per step: gates[16,768] = [x_t | h](16x512) @ W(512x768) via
// bf16 WMMA / f32 accumulate; gate-h keeps x-part and h-part separate
// (reset_after GRU). Weight fragments double-buffered and streamed from L2
// each step; LICM across t is blocked by an opaque *offset* (keeps global
// address-space provenance -> global_load_b128, not flat_load). x_t is
// gathered straight into LDS with global_load_async_to_lds_b128.
// ---------------------------------------------------------------------------
__global__ void __launch_bounds__(256) __attribute__((amdgpu_waves_per_eu(2)))
gru_recur_kernel(const long long* __restrict__ ids,
                 const int* __restrict__ lengths,
                 const unsigned int* __restrict__ embp,
                 const unsigned int* __restrict__ wp,
                 const float* __restrict__ b_ih,
                 const float* __restrict__ b_hh,
                 unsigned int* __restrict__ h_out) {
  __shared__ unsigned int sh_x[16 * 128];   // x_t, bf16 pairs, row-major
  __shared__ unsigned int sh_h[16 * 128];   // h,   bf16 pairs, row-major

  const int tid  = threadIdx.x;
  const int lane = tid & 31;
  const int wave = tid >> 5;
  const int half = lane >> 4;
  const int nl   = lane & 15;
  const int b0   = blockIdx.x * 16;
  const int c0   = wave * 2;

  for (int i = tid; i < 16 * 128; i += 256) sh_h[i] = 0u;

  int len_r[8];
#pragma unroll
  for (int e = 0; e < 8; ++e) len_r[e] = lengths[b0 + half * 8 + e];
  float bsz[2], bsr[2], bxh[2], brh[2];
#pragma unroll
  for (int ct = 0; ct < 2; ++ct) {
    int j = (c0 + ct) * 16 + nl;
    bsz[ct] = b_ih[j] + b_hh[j];
    bsr[ct] = b_ih[HH + j] + b_hh[HH + j];
    bxh[ct] = b_ih[2 * HH + j];
    brh[ct] = b_hh[2 * HH + j];
  }
  float hreg[2][8] = {};

  const v8f vzero = {0.f, 0.f, 0.f, 0.f, 0.f, 0.f, 0.f, 0.f};

  for (int t = 0; t < TT; ++t) {
    // ---- gather x_t = emb[ids[:, t]] into LDS via async-to-LDS b128 ----
#pragma unroll
    for (int i = 0; i < 2; ++i) {
      int idx = tid + i * 256;             // 0..511 : 16 rows x 32 16B-chunks
      int bb = idx >> 5, cc = idx & 31;
      long long id = ids[(long long)(b0 + bb) * TT + t];
      const unsigned int* g = embp + (int)id * 128 + cc * 4;
      unsigned int lds_off = (unsigned int)(uintptr_t)&sh_x[bb * 128 + cc * 4];
      asm volatile("global_load_async_to_lds_b128 %0, %1, off"
                   :: "v"(lds_off), "v"(g) : "memory");
    }
    asm volatile("s_wait_asynccnt 0x0" ::: "memory");
    __syncthreads();

    // ---- GEMM: K = 512 (x over k<256, h over k>=256) ----
    v8f accz[2]  = {vzero, vzero};
    v8f accr[2]  = {vzero, vzero};
    v8f accxh[2] = {vzero, vzero};
    v8f accrh[2] = {vzero, vzero};

    // Opaque per-step *offset*: blocks LICM from hoisting the (t-invariant)
    // weight loads out of the 200-step loop, while preserving the global
    // address-space provenance of wp (emits global_load_b128, not flat_load).
    unsigned int toff = 0u;
    asm volatile("" : "+s"(toff));
    const unsigned int* wpt = wp + toff;

    Frag fb0[6], fb1[6], fa;
    load_b6(fb0, wpt, 0, c0, lane);
#pragma unroll
    for (int kt2 = 0; kt2 < 8; ++kt2) {
      const int kt = kt2 * 2;
      load_b6(fb1, wpt, kt + 1, c0, lane);                 // prefetch k+1
      load_a(fa, (kt < 8) ? sh_x : sh_h, nl, half, kt & 7);
      wmma6(fa, fb0, kt, accz, accr, accxh, accrh);
      if (kt2 < 7) load_b6(fb0, wpt, kt + 2, c0, lane);    // prefetch k+2
      load_a(fa, (kt + 1 < 8) ? sh_x : sh_h, nl, half, (kt + 1) & 7);
      wmma6(fa, fb1, kt + 1, accz, accr, accxh, accrh);
    }
    __syncthreads();   // all reads of sh_x/sh_h done

    // ---- GRU elementwise update in registers; rebroadcast h as bf16 ----
#pragma unroll
    for (int ct = 0; ct < 2; ++ct) {
      int j = (c0 + ct) * 16 + nl;
#pragma unroll
      for (int e = 0; e < 8; ++e) {
        int m = half * 8 + e;
        if (t < len_r[e]) {
          float z  = sigmoid_fast(accz[ct][e] + bsz[ct]);
          float r  = sigmoid_fast(accr[ct][e] + bsr[ct]);
          float hh = tanh_fast(accxh[ct][e] + bxh[ct] +
                               r * (accrh[ct][e] + brh[ct]));
          hreg[ct][e] = z * hreg[ct][e] + (1.f - z) * hh;
        }
        ((unsigned short*)sh_h)[m * 256 + j] = f2bf(hreg[ct][e]);
      }
    }
    __syncthreads();   // sh_h ready for next step's A fragments
  }

  // ---- write h_final (bf16 pairs) ----
  for (int i = tid; i < 2048; i += 256) {
    int m = i >> 7, c = i & 127;
    h_out[(b0 + m) * 128 + c] = sh_h[m * 128 + c];
  }
}

// ---------------------------------------------------------------------------
// logits = h_final(256x256) @ emb^T(256x50001).  emb row-major == B-fragment
// contiguous.  3126 blocks x 128 thr: block = one 16-col N-tile; wave w
// handles M-tiles 4w..4w+3.
// ---------------------------------------------------------------------------
__global__ void __launch_bounds__(128)
logits_kernel(const unsigned int* __restrict__ hp,
              const unsigned int* __restrict__ embp,
              float* __restrict__ out) {
  const int lane = threadIdx.x & 31;
  const int wave = threadIdx.x >> 5;
  const int half = lane >> 4;
  const int nl   = lane & 15;
  const int nt   = blockIdx.x;
  const int n    = nt * 16 + nl;
  const int row  = (n > VV) ? VV : n;   // clamp OOB fragment loads

  const v8f vzero = {0.f, 0.f, 0.f, 0.f, 0.f, 0.f, 0.f, 0.f};
  v8f acc[4] = {vzero, vzero, vzero, vzero};

  for (int kt = 0; kt < 8; ++kt) {       // K = 256
    Frag fb;                              // B = emb^T fragment: contiguous 32B
    const u4_t* bp = (const u4_t*)(embp + row * 128 + kt * 16 + half * 8);
    fb.q[0] = bp[0]; fb.q[1] = bp[1];
#pragma unroll
    for (int mt = 0; mt < 4; ++mt) {
      int m = (wave * 4 + mt) * 16 + nl;
      Frag fa;
      int base = m * 128 + kt * 16 + half * 8;
#pragma unroll
      for (int d = 0; d < 4; ++d) fa.u[d]     = hp[base + d];
#pragma unroll
      for (int d = 0; d < 4; ++d) fa.u[4 + d] = hp[base + 4 + d];
      acc[mt] = wmma_bf16(fa, fb, acc[mt]);
    }
  }

#pragma unroll
  for (int mt = 0; mt < 4; ++mt) {
#pragma unroll
    for (int e = 0; e < 8; ++e) {
      int m = (wave * 4 + mt) * 16 + half * 8 + e;
      if (n <= VV) out[(size_t)m * (VV + 1) + n] = acc[mt][e];
    }
  }
}

// ---------------------------------------------------------------------------
extern "C" void kernel_launch(void* const* d_in, const int* in_sizes, int n_in,
                              void* d_out, int out_size, void* d_ws, size_t ws_size,
                              hipStream_t stream) {
  const long long* ids     = (const long long*)d_in[0];  // int64 [B,T]
  const int*       lengths = (const int*)d_in[1];        // int32 [B]
  const float*     emb     = (const float*)d_in[2];      // [V+1, H]
  const float*     w_ih    = (const float*)d_in[3];      // [H, 3H]
  const float*     w_hh    = (const float*)d_in[4];      // [H, 3H]
  const float*     b_ih    = (const float*)d_in[5];      // [3H]
  const float*     b_hh    = (const float*)d_in[6];      // [3H]
  float* out = (float*)d_out;                            // [B, V+1] f32

  unsigned int* ws   = (unsigned int*)d_ws;
  unsigned int* embp = ws;                    // EMB_DW dwords  (~25.6 MB)
  unsigned int* wp   = ws + EMB_DW;           // WP_DW dwords   (~0.79 MB)
  unsigned int* hp   = wp + WP_DW;            // 32768 dwords   (0.13 MB)

  pack_emb_kernel<<<(EMB_DW + 255) / 256, 256, 0, stream>>>(emb, embp);
  pack_w_kernel<<<WP_DW / 256, 256, 0, stream>>>(w_ih, w_hh, wp);
  gru_recur_kernel<<<16, 256, 0, stream>>>(ids, lengths, embp, wp, b_ih, b_hh, hp);
  logits_kernel<<<(VV + 16) / 16, 128, 0, stream>>>(hp, embp, out);
}
